// Joiner_14594298872273
// MI455X (gfx1250) — matrix operand order repaired and checked by reading, not compiled, not run on hardware
//
#include <hip/hip_runtime.h>

typedef __attribute__((ext_vector_type(4)))  __bf16 bf16x4;
typedef __attribute__((ext_vector_type(8)))  __bf16 bf16x8;
typedef __attribute__((ext_vector_type(16))) __bf16 bf16x16;
typedef __attribute__((ext_vector_type(8)))  float  v8f;

union Frag16 { bf16x16 v; bf16x8 h[2]; };

// ---- constants from the reference problem ----
// N=16, T=512, U=64, D=512, V=500
#define DIM_T 512
#define DIM_U 64
#define DIM_D 512
#define DIM_V 500
#define LDA   520   // padded LDS row stride (bf16 elems): 1040B -> rotates 4 banks/row
#define MROWS 64    // M rows per block (4 WMMA M-tiles)

// Convert W (V x D f32, row-major) -> bf16 (512 x 512), zero-padding rows V..511.
__global__ __launch_bounds__(256) void wcvt_kernel(const float* __restrict__ W,
                                                   __bf16* __restrict__ Wb) {
    int i = blockIdx.x * 256 + threadIdx.x;     // 0 .. 512*512-1
    int n = i >> 9;
    int k = i & 511;
    float v = (n < DIM_V) ? W[n * DIM_D + k] : 0.0f;
    Wb[i] = (__bf16)v;
}

__global__ __launch_bounds__(256) void joiner_kernel(
    const float* __restrict__ enc,      // (16, T, D)
    const float* __restrict__ dec,      // (16, U, D)
    const __bf16* __restrict__ Wb,      // (512, 512) bf16, row-major [n][k]
    const float* __restrict__ bias,     // (V,)
    const int* __restrict__ bI,
    const int* __restrict__ tI,
    const int* __restrict__ uI,
    float* __restrict__ out,            // (M, V)
    int M) {
    __shared__ __bf16 ldsA[MROWS * LDA];   // 64 activation rows, all of K, bf16

    const int tid       = threadIdx.x;
    const int blockBase = blockIdx.x * MROWS;

    // -------- Stage 1: ragged gather + add + tanh -> bf16 rows in LDS --------
    {
        const int c = tid & 15;              // 16 threads per row
#pragma unroll
        for (int g = 0; g < 4; ++g) {        // 4 row groups of 16
            const int r = (tid >> 4) + 16 * g;
            const int pair = blockBase + r;
            __bf16* rowp = &ldsA[r * LDA];
            if (pair < M) {
                const int b = bI[pair], t = tI[pair], u = uI[pair];
                const float4* e4 = (const float4*)(enc + ((size_t)b * DIM_T + t) * DIM_D);
                const float4* d4 = (const float4*)(dec + ((size_t)b * DIM_U + u) * DIM_D);
#pragma unroll
                for (int j = 0; j < 8; ++j) {
                    const int c4 = c + 16 * j;       // float4 index within row
                    float4 ev = e4[c4];
                    float4 dv = d4[c4];
                    bf16x4 pk;
                    pk[0] = (__bf16)tanhf(ev.x + dv.x);
                    pk[1] = (__bf16)tanhf(ev.y + dv.y);
                    pk[2] = (__bf16)tanhf(ev.z + dv.z);
                    pk[3] = (__bf16)tanhf(ev.w + dv.w);
                    *(bf16x4*)(rowp + c4 * 4) = pk;
                }
            } else {
                bf16x4 z = {};
#pragma unroll
                for (int j = 0; j < 8; ++j)
                    *(bf16x4*)(rowp + (c + 16 * j) * 4) = z;
            }
        }
    }
    __syncthreads();

    // -------- Stage 2: 8 waves, each 64M x 64N strip (4x4 WMMA tiles), K=512 --------
    const int lane   = tid & 31;
    const int wave   = tid >> 5;
    const int laneLo = lane & 15;
    const int half   = lane >> 4;        // 0 or 1
    const int khalfA = half << 4;        // A: K-run starts at 0 or 16
    const int khalfB = half << 4;        // B: K-run starts at 0 or 16
    const int nBase  = wave * 64;        // 8 waves cover 512 padded columns

    v8f acc[4][4] = {};                  // [m-tile][n-tile], 128 VGPRs

    for (int kc = 0; kc < DIM_D; kc += 32) {
        // A fragments for the 4 M-tiles (16-bit A 16x32: lane half holds contiguous K run of 16)
        Frag16 a[4];
#pragma unroll
        for (int mt = 0; mt < 4; ++mt) {
            const bf16x8* ap =
                (const bf16x8*)(&ldsA[(mt * 16 + laneLo) * LDA + kc + khalfA]);
            a[mt].h[0] = ap[0];
            a[mt].h[1] = ap[1];
        }
#pragma unroll
        for (int tt = 0; tt < 4; ++tt) {
            // B fragment (16-bit B 32x16): lane half holds contiguous K run of 16
            const int n = nBase + tt * 16 + laneLo;
            bf16x16 bm = *(const bf16x16*)(Wb + (size_t)n * DIM_D + kc + khalfB);
#pragma unroll
            for (int mt = 0; mt < 4; ++mt) {
                acc[mt][tt] = __builtin_amdgcn_wmma_f32_16x16x32_bf16(
                    false, a[mt].v, false, bm, (short)0, acc[mt][tt], false, false);
            }
        }
    }

    // -------- Epilogue: bias + guarded stores (f32 C/D 16x16 layout) --------
#pragma unroll
    for (int tt = 0; tt < 4; ++tt) {
        const int n = nBase + tt * 16 + laneLo;
        const bool nok = (n < DIM_V);
        const float bv = nok ? bias[n] : 0.0f;
#pragma unroll
        for (int mt = 0; mt < 4; ++mt) {
#pragma unroll
            for (int r = 0; r < 8; ++r) {
                const int mrow = blockBase + mt * 16 + (half ? (8 + r) : r);
                if (nok && mrow < M)
                    out[(size_t)mrow * DIM_V + n] = acc[mt][tt][r] + bv;
            }
        }
    }
}

extern "C" void kernel_launch(void* const* d_in, const int* in_sizes, int n_in,
                              void* d_out, int out_size, void* d_ws, size_t ws_size,
                              hipStream_t stream) {
    const float* enc  = (const float*)d_in[0];
    const float* dec  = (const float*)d_in[1];
    const float* W    = (const float*)d_in[2];
    const float* bias = (const float*)d_in[3];
    const int*   bI   = (const int*)d_in[4];
    const int*   tI   = (const int*)d_in[5];
    const int*   uI   = (const int*)d_in[6];
    float* out = (float*)d_out;
    const int M = in_sizes[4];          // total ragged (b,t,u) pairs

    __bf16* Wb = (__bf16*)d_ws;         // 512*512*2 = 512 KB scratch

    wcvt_kernel<<<(512 * 512) / 256, 256, 0, stream>>>(W, Wb);

    const int mtiles = (M + MROWS - 1) / MROWS;
    if (mtiles > 0) {
        joiner_kernel<<<mtiles, 256, 0, stream>>>(enc, dec, Wb, bias, bI, tI, uI, out, M);
    }
}
